// ConnectLoss_57629871178300
// MI455X (gfx1250) — matrix operand order repaired and checked by reading, not compiled, not run on hardware
//
#include <hip/hip_runtime.h>
#include <hip/hip_bf16.h>
#include <math.h>

// ---------------------------------------------------------------------------
// Problem constants (B, C, H, W) = (8, 1, 512, 512), C folded away.
// ---------------------------------------------------------------------------
#define BB   8
#define HH   512
#define WW   512
#define HWp  (HH * WW)           // 262144
#define NEG_CLAMP (-100.0f)
#define LDW  40                  // LDS row stride in halves (80 B, 16B-aligned)

// Workspace layout (bytes).
constexpr size_t ACC_OFF    = 0;                        // 128 floats reserved
constexpr size_t SIGH_OFF   = 1024;                     // sigmoid(pred) as f16
constexpr size_t SIGH_BYTES = (size_t)BB * 8 * HWp * 2; // 33.5 MB
constexpr size_t HORI_OFF   = SIGH_OFF + SIGH_BYTES;    // hori as f16
constexpr size_t TRN_BYTES  = (size_t)BB * HWp * 2;     // 4.2 MB each
constexpr size_t VERT_OFF   = HORI_OFF + TRN_BYTES;     // verti as f16
constexpr size_t SBUF_OFF   = VERT_OFF + TRN_BYTES;     // stage-1 intermediates (f16)
constexpr size_t SBUF_BYTES = (size_t)BB * 4 * HWp * 2; // 16.8 MB
constexpr size_t RBUF_OFF   = SBUF_OFF + SBUF_BYTES;    // voting results (f32), 67 MB

// Accumulator slots (floats)
#define ACC_BCE   0
#define ACC_ENUM  1
#define ACC_EDEN  2
#define ACC_INTER 3
#define ACC_FPS   11
#define ACC_TS    19

typedef __attribute__((ext_vector_type(16))) _Float16 v16h;
typedef __attribute__((ext_vector_type(8)))  float    v8f;

union FragH {
    v16h         v;
    _Float16     e[16];
    unsigned int u[8];
};

// ---------------------------------------------------------------------------
// CDNA5 async global->LDS copy (GLOBAL_LOAD_ASYNC_TO_LDS_B128, ASYNCcnt).
// Builtin signature (from clang diagnostic): param0 = AS1 int-vec4*,
// param1 = AS3 int-vec4*, then imm offset + imm cpol.
// ---------------------------------------------------------------------------
typedef int v4i_vs __attribute__((__vector_size__(16)));
typedef __attribute__((address_space(1))) v4i_vs* gv4i_p;  // global ptr
typedef __attribute__((address_space(3))) v4i_vs* lv4i_p;  // LDS ptr

__device__ __forceinline__ void async_copy_b128(void* lds, const void* g) {
#if __has_builtin(__builtin_amdgcn_global_load_async_to_lds_b128)
    __builtin_amdgcn_global_load_async_to_lds_b128(
        (gv4i_p)(uintptr_t)g,
        (lv4i_p)(unsigned)(uintptr_t)lds,   // low 32 bits = LDS byte offset
        0, 0);
#else
    unsigned           loff = (unsigned)(uintptr_t)lds;
    unsigned long long ga   = (unsigned long long)(uintptr_t)g;
    asm volatile("global_load_async_to_lds_b128 %0, %1, off"
                 :: "v"(loff), "v"(ga) : "memory");
#endif
}

__device__ __forceinline__ void wait_asynccnt0() {
#if __has_builtin(__builtin_amdgcn_s_wait_asynccnt)
    __builtin_amdgcn_s_wait_asynccnt(0);
#else
    asm volatile("s_wait_asynccnt 0" ::: "memory");
#endif
}

// ---------------------------------------------------------------------------
// Kernel 0: zero the scalar accumulators (graph replays need clean slate).
// ---------------------------------------------------------------------------
__global__ void k_init_acc(float* acc) {
    acc[threadIdx.x] = 0.0f;
}

// ---------------------------------------------------------------------------
// Kernel 1: f32 -> f16 conversion of translation matrices.
// ---------------------------------------------------------------------------
__global__ void k_convert_f16(const float* __restrict__ hori,
                              const float* __restrict__ verti,
                              _Float16* __restrict__ horih,
                              _Float16* __restrict__ vertih) {
    size_t i = (size_t)blockIdx.x * blockDim.x + threadIdx.x;
    horih[i]  = (_Float16)hori[i];
    vertih[i] = (_Float16)verti[i];
}

// ---------------------------------------------------------------------------
// Block reduction helper (blockDim.x == 256).
// ---------------------------------------------------------------------------
__device__ __forceinline__ float block_reduce(float v, float* red, int tid) {
    red[tid] = v;
    __syncthreads();
    for (int s = 128; s > 0; s >>= 1) {
        if (tid < s) red[tid] += red[tid + s];
        __syncthreads();
    }
    float r = red[0];
    __syncthreads();
    return r;
}

// ---------------------------------------------------------------------------
// Kernel 2: fused elementwise pass (conn_t, sigmoid, BCE, edge terms,
// per-b target sums, f16 sigmoid stash). One thread per (b,h,w).
// ---------------------------------------------------------------------------
__global__ __launch_bounds__(256)
void k_elementwise(const float* __restrict__ pred,
                   const float* __restrict__ target,
                   _Float16* __restrict__ sigh,
                   float* __restrict__ acc) {
    __shared__ float red[256];
    const int tid = threadIdx.x;
    const size_t idx = (size_t)blockIdx.x * 256 + tid;   // [0, B*HW)
    const int b   = (int)(idx >> 18);                    // HW = 2^18
    const int rem = (int)(idx & (HWp - 1));
    const int h   = rem >> 9;
    const int w   = rem & (WW - 1);

    const float* tb = target + (size_t)b * HWp;
    const float  m  = tb[rem];

    auto nb = [&](int dh, int dw) -> float {
        int hh = h + dh, ww = w + dw;
        if (hh < 0 || hh >= HH || ww < 0 || ww >= WW) return 0.0f;
        return tb[hh * WW + ww];
    };
    float conn[8];
    conn[0] = m * nb(-1, -1);  // m * down_right
    conn[1] = m * nb(-1,  0);  // m * down
    conn[2] = m * nb(-1, +1);  // m * down_left
    conn[3] = m * nb( 0, -1);  // m * right
    conn[4] = m * nb( 0, +1);  // m * left
    conn[5] = m * nb(+1, -1);  // m * up_right
    conn[6] = m * nb(+1,  0);  // m * up
    conn[7] = m * nb(+1, +1);  // m * up_left

    float bce = 0.0f, pmin = 1.0f, sumc = 0.0f;
#pragma unroll
    for (int k = 0; k < 8; ++k) {
        float x = pred[((size_t)b * 8 + k) * HWp + rem];
        float p = 1.0f / (1.0f + __expf(-x));
        sigh[((size_t)b * 8 + k) * HWp + rem] = (_Float16)p;
        float t   = conn[k];
        float lp  = fmaxf(logf(p), NEG_CLAMP);
        float l1p = fmaxf(log1pf(-p), NEG_CLAMP);
        bce  += -(t * lp + (1.0f - t) * l1p);
        pmin  = fminf(pmin, p);
        sumc += t;
    }
    float edge  = (sumc > 0.0f && sumc < 8.0f) ? 1.0f : 0.0f;
    float pm    = pmin * edge;
    float enum_ = -fmaxf(log1pf(-pm), NEG_CLAMP);

    float rb = block_reduce(bce, red, tid);
    if (tid == 0) atomicAdd(&acc[ACC_BCE], rb);
    float rn = block_reduce(enum_, red, tid);
    if (tid == 0) atomicAdd(&acc[ACC_ENUM], rn);
    float rd = block_reduce(pm, red, tid);
    if (tid == 0) atomicAdd(&acc[ACC_EDEN], rd);
    float rt = block_reduce(m, red, tid);
    if (tid == 0) atomicAdd(&acc[ACC_TS + b], rt);
}

// ---------------------------------------------------------------------------
// Kernel 3: WMMA GEMM with async global->LDS staging.
// Workgroup = 128 threads (4 waves) -> 32x32 output block (16x16 per wave).
// Per K-step: stage 32x32 f16 tiles of the A/B source matrices into LDS with
// coalesced b128 async copies (transpose absorbed into LDS indexing), then
// each wave builds its WMMA fragments from LDS and issues
// v_wmma_f32_16x16x32_f16.
//   stage 1 (z=0..7):  right/left/bottom/up finals + 4 verti intermediates
//   stage 2 (z=0..3):  hori pass over the intermediates
// ---------------------------------------------------------------------------
__global__ __launch_bounds__(128)
void k_gemm(const _Float16* __restrict__ sigh,
            const _Float16* __restrict__ horih,
            const _Float16* __restrict__ vertih,
            _Float16* __restrict__ sbuf,
            float* __restrict__ rbuf,
            int stage) {
    __shared__ _Float16 At[32 * LDW];
    __shared__ _Float16 Bt[32 * LDW];

    const int tid  = threadIdx.x;
    const int lane = tid & 31;
    const int wv   = tid >> 5;            // wave 0..3
    const int wi   = wv >> 1, wj = wv & 1;
    const int bm   = blockIdx.x >> 4, bn = blockIdx.x & 15;
    const int b    = blockIdx.y;
    const int z    = blockIdx.z;
    const int g    = (lane >> 4) & 1;
    const int l15  = lane & 15;
    const int rowBlock = bm * 32, colBlock = bn * 32;

    const _Float16* hb = horih + (size_t)b * HWp;
    const _Float16* vb = vertih + (size_t)b * HWp;
    const _Float16* sg = sigh + (size_t)b * 8 * HWp;
    _Float16* sb = sbuf + (size_t)b * 4 * HWp;
    float*    Rb = rbuf + (size_t)b * 8 * HWp;

    const _Float16* A;  bool Acol;   // Acol: A = P^T (fragment reads transposed)
    const _Float16* Bm; bool Brow;   // Brow: B = P^T
    float* outF = nullptr; _Float16* outH = nullptr;

    if (stage == 1) {
        switch (z) {
        case 0: A = sg + 4 * HWp; Acol = false; Bm = hb; Brow = false; outF = Rb + 3 * HWp; break; // right = c4 @ hori
        case 1: A = sg + 3 * HWp; Acol = false; Bm = hb; Brow = true;  outF = Rb + 4 * HWp; break; // left = c3 @ hori^T
        case 2: A = vb; Acol = true;  Bm = sg + 6 * HWp; Brow = false; outF = Rb + 1 * HWp; break; // bottom = v^T @ c6
        case 3: A = vb; Acol = false; Bm = sg + 1 * HWp; Brow = false; outF = Rb + 6 * HWp; break; // up = v @ c1
        case 4: A = vb; Acol = false; Bm = sg + 0 * HWp; Brow = false; outH = sb + 0 * HWp; break; // s0 = v @ c0
        case 5: A = vb; Acol = false; Bm = sg + 2 * HWp; Brow = false; outH = sb + 1 * HWp; break; // s2 = v @ c2
        case 6: A = vb; Acol = true;  Bm = sg + 5 * HWp; Brow = false; outH = sb + 2 * HWp; break; // s5 = v^T @ c5
        default:A = vb; Acol = true;  Bm = sg + 7 * HWp; Brow = false; outH = sb + 3 * HWp; break; // s7 = v^T @ c7
        }
    } else {
        switch (z) {
        case 0: A = sb + 3 * HWp; Acol = false; Bm = hb; Brow = false; outF = Rb + 0 * HWp; break; // right_bottom = s7 @ hori
        case 1: A = sb + 2 * HWp; Acol = false; Bm = hb; Brow = true;  outF = Rb + 2 * HWp; break; // left_bottom = s5 @ hori^T
        case 2: A = sb + 1 * HWp; Acol = false; Bm = hb; Brow = false; outF = Rb + 5 * HWp; break; // right_above = s2 @ hori
        default:A = sb + 0 * HWp; Acol = false; Bm = hb; Brow = true;  outF = Rb + 7 * HWp; break; // left_above = s0 @ hori^T
        }
    }

    // staging geometry: each thread copies one 16 B chunk per tile per K-step
    const int sr = tid >> 2;              // 0..31 (tile row)
    const int sc = (tid & 3) * 8;         // 0,8,16,24 (halves)
    const int iL = wi * 16 + l15;         // A fragment row within 32-block
    const int nL = wj * 16 + l15;         // B fragment col within 32-block

    v8f acc = {};
    for (int k0 = 0; k0 < WW; k0 += 32) {
        // 32x32 source-matrix tiles; transpose handled by LDS indexing below
        const int ra0 = Acol ? k0 : rowBlock;
        const int ca0 = Acol ? rowBlock : k0;
        const int rb0 = Brow ? colBlock : k0;
        const int cb0 = Brow ? k0 : colBlock;

        async_copy_b128(&At[sr * LDW + sc], A  + (size_t)(ra0 + sr) * WW + ca0 + sc);
        async_copy_b128(&Bt[sr * LDW + sc], Bm + (size_t)(rb0 + sr) * WW + cb0 + sc);
        if (k0 + 32 < WW)   // speculative prefetch of next A tile row (L2)
            __builtin_prefetch(A + (size_t)(ra0 + sr) * WW + ca0 + sc + 32, 0, 1);
        wait_asynccnt0();
        __syncthreads();

        // A fragment: half j -> K = g*16 + j
        FragH a, bf;
        if (!Acol) {
#pragma unroll
            for (int e = 0; e < 8; ++e)
                a.u[e] = *(const unsigned int*)&At[iL * LDW + g * 16 + 2 * e];
        } else {
#pragma unroll
            for (int j = 0; j < 16; ++j) {
                int k = g * 16 + j;
                a.e[j] = At[k * LDW + iL];
            }
        }
        // B fragment: half j -> K = g*16 + j
        if (!Brow) {
#pragma unroll
            for (int j = 0; j < 16; ++j)
                bf.e[j] = Bt[(g * 16 + j) * LDW + nL];
        } else {
#pragma unroll
            for (int e = 0; e < 8; ++e)
                bf.u[e] = *(const unsigned int*)&Bt[nL * LDW + g * 16 + 2 * e];
        }

        acc = __builtin_amdgcn_wmma_f32_16x16x32_f16(
            /*neg_a=*/false, a.v, /*neg_b=*/false, bf.v,
            /*c_mod=*/(short)0, acc, /*reuse_a=*/false, /*reuse_b=*/false);

        __syncthreads();   // protect LDS tiles before next staging round
    }

    const int n  = colBlock + wj * 16 + l15;
    const int mb = rowBlock + wi * 16 + g * 8;
    if (outF) {
#pragma unroll
        for (int r = 0; r < 8; ++r)
            outF[(size_t)(mb + r) * WW + n] = acc[r];
    } else {
#pragma unroll
        for (int r = 0; r < 8; ++r)
            outH[(size_t)(mb + r) * WW + n] = (_Float16)acc[r];
    }
}

// ---------------------------------------------------------------------------
// Kernel 4: voting max + dice partial sums.
// ---------------------------------------------------------------------------
__global__ __launch_bounds__(256)
void k_vote(const _Float16* __restrict__ sigh,
            const float* __restrict__ rbuf,
            const float* __restrict__ target,
            float* __restrict__ acc) {
    __shared__ float red[256];
    const int tid = threadIdx.x;
    const size_t idx = (size_t)blockIdx.x * 256 + tid;
    const int b   = (int)(idx >> 18);
    const int rem = (int)(idx & (HWp - 1));

    float fin = -1.0e30f;
#pragma unroll
    for (int k = 0; k < 8; ++k) {
        float p = (float)sigh[((size_t)b * 8 + k) * HWp + rem];
        float r = rbuf[((size_t)b * 8 + k) * HWp + rem];
        fin = fmaxf(fin, p * r);
    }
    float m = target[(size_t)b * HWp + rem];

    float ri = block_reduce(fin * m, red, tid);
    if (tid == 0) atomicAdd(&acc[ACC_INTER + b], ri);
    float rf = block_reduce(fin, red, tid);
    if (tid == 0) atomicAdd(&acc[ACC_FPS + b], rf);
}

// ---------------------------------------------------------------------------
// Kernel 5: finalize scalar loss.
// ---------------------------------------------------------------------------
__global__ void k_finalize(const float* __restrict__ acc, float* __restrict__ out) {
    float conn_loss = acc[ACC_BCE] / (float)((size_t)BB * 8 * HWp);
    float edge_loss = acc[ACC_ENUM] / acc[ACC_EDEN];
    float seg = 0.0f;
    for (int b = 0; b < BB; ++b) {
        float inter = acc[ACC_INTER + b];
        float uni   = acc[ACC_FPS + b] + acc[ACC_TS + b];
        float dice  = (2.0f * inter + 1.0f) / (uni + 1.0f);
        seg += (1.0f - dice);
    }
    seg /= (float)BB;
    out[0] = conn_loss + edge_loss + seg;
}

// ---------------------------------------------------------------------------
// Host launcher. Inputs (setup_inputs order): pred, target, hori, verti.
// ---------------------------------------------------------------------------
extern "C" void kernel_launch(void* const* d_in, const int* in_sizes, int n_in,
                              void* d_out, int out_size, void* d_ws, size_t ws_size,
                              hipStream_t stream) {
    const float* pred   = (const float*)d_in[0];
    const float* target = (const float*)d_in[1];
    const float* hori   = (const float*)d_in[2];
    const float* verti  = (const float*)d_in[3];

    char* ws = (char*)d_ws;
    float*     acc    = (float*)(ws + ACC_OFF);
    _Float16*  sigh   = (_Float16*)(ws + SIGH_OFF);
    _Float16*  horih  = (_Float16*)(ws + HORI_OFF);
    _Float16*  vertih = (_Float16*)(ws + VERT_OFF);
    _Float16*  sbuf   = (_Float16*)(ws + SBUF_OFF);
    float*     rbuf   = (float*)(ws + RBUF_OFF);

    const int nBHW_blocks = (BB * HWp) / 256;     // 8192

    k_init_acc<<<1, 128, 0, stream>>>(acc);
    k_convert_f16<<<(BB * HWp) / 256, 256, 0, stream>>>(hori, verti, horih, vertih);
    k_elementwise<<<nBHW_blocks, 256, 0, stream>>>(pred, target, sigh, acc);
    k_gemm<<<dim3(256, BB, 8), 128, 0, stream>>>(sigh, horih, vertih, sbuf, rbuf, 1);
    k_gemm<<<dim3(256, BB, 4), 128, 0, stream>>>(sigh, horih, vertih, sbuf, rbuf, 2);
    k_vote<<<nBHW_blocks, 256, 0, stream>>>(sigh, rbuf, target, acc);
    k_finalize<<<1, 1, 0, stream>>>(acc, (float*)d_out);
}
